// kernel_generated_3_62904091017253
// MI455X (gfx1250) — compile-verified, matrix operand order checked by reading, not run on hardware
//
#include <hip/hip_runtime.h>
#include <hip/hip_bf16.h>

// ---------------------------------------------------------------------------
// Fused: t4 = im2col-GEMM (M=196 per image, K=384, N=64 per c-chunk) in bf16
// hi/lo split (3 WMMA products ~ fp32 accuracy), t4 staged in LDS, then the
// rolled-w1 epilogue:  out[m,c,a,b] = t4[(a-1)%14,b,c]*w1[c,0]
//                                   + t4[(a-2)%14,b,c]*w1[c,1]
// ---------------------------------------------------------------------------

typedef __bf16  bf16_t;
typedef __attribute__((ext_vector_type(16))) __bf16        v16bf;
typedef __attribute__((ext_vector_type(8)))  float         v8f;
typedef __attribute__((ext_vector_type(4)))  unsigned int  v4u;

// LDS geometry (elements)
#define XS_ELEMS   (14 * 16 * 136)   // xsT[l][hh][jpad] bf16, hh has halo
#define BS_STRIDE  392               // padded K stride (bank-conflict-free)
#define BS_ELEMS   (64 * BS_STRIDE)  // Bs[n][kk] bf16
#define TS_STRIDE  68                // padded t4 row stride (floats)
#define TS_ELEMS   (196 * TS_STRIDE)
#define SMEM_BYTES ((XS_ELEMS * 2 + BS_ELEMS * 2) * 2 * 2 + TS_ELEMS * 4) // 497728

static __device__ __forceinline__ unsigned short f2bf_rne(float f) {
    unsigned int u = __float_as_uint(f);
    return (unsigned short)((u + 0x7FFFu + ((u >> 16) & 1u)) >> 16);
}
static __device__ __forceinline__ float bf2f(unsigned short h) {
    return __uint_as_float(((unsigned int)h) << 16);
}
static __device__ __forceinline__ v16bf ld_frag(const unsigned short* p0,
                                                const unsigned short* p1) {
    union { v16bf bf; v4u u[2]; } f;
    f.u[0] = *(const v4u*)p0;   // K = base .. base+7   (16B, aligned)
    f.u[1] = *(const v4u*)p1;   // K = base+16 .. +23
    return f.bf;
}

__global__ __launch_bounds__(256)
void fused_conv_roll_wmma(const float* __restrict__ x,
                          const float* __restrict__ w1,
                          const float* __restrict__ w2,
                          float* __restrict__ out) {
    extern __shared__ unsigned char smem[];
    unsigned short* xs_hi = (unsigned short*)smem;           // 30464 bf16
    unsigned short* xs_lo = xs_hi + XS_ELEMS;
    unsigned short* bs_hi = xs_lo + XS_ELEMS;                // 25088 bf16
    unsigned short* bs_lo = bs_hi + BS_ELEMS;
    float*          ts    = (float*)(bs_lo + BS_ELEMS);      // 196x68 f32

    const int m   = blockIdx.y;
    const int c0  = blockIdx.x * 64;
    const int tid = threadIdx.x;

    // ---- 1) zero the h-halo planes (hh == 0 and hh == 15) of xsT ----------
    for (int i = tid; i < 14 * 136; i += 256) {
        int l = i / 136, j = i - l * 136;
        int a0 = (l * 16 + 0) * 136 + j;
        int a1 = (l * 16 + 15) * 136 + j;
        xs_hi[a0] = 0; xs_lo[a0] = 0;
        xs_hi[a1] = 0; xs_lo[a1] = 0;
    }

    // ---- 2) load x[m] (128 x 14 x 14), split hi/lo bf16, transpose to -----
    //         xsT[l][h+1][j]
    const float* xm = x + (size_t)m * 25088;
    for (int i = tid; i < 25088; i += 256) {
        int j = i / 196; int rem = i - j * 196;
        int l = rem / 14; int h = rem - l * 14;
        float v = xm[i];
        unsigned short rh = f2bf_rne(v);
        float fl = v - bf2f(rh);
        unsigned short rl = f2bf_rne(fl);
        int addr = (l * 16 + h + 1) * 136 + j;
        xs_hi[addr] = rh;
        xs_lo[addr] = rl;
    }

    // ---- 3) load w2 slice -> Bs[n][kk], kk = tap*128 + j (K-major) --------
    for (int i = tid; i < 64 * 384; i += 256) {
        int kk = i >> 6; int n = i & 63;          // n fastest -> coalesced
        int tap = kk >> 7; int j = kk & 127;
        float v = w2[(j * 3 + tap) * 256 + c0 + n];
        unsigned short rh = f2bf_rne(v);
        float fl = v - bf2f(rh);
        unsigned short rl = f2bf_rne(fl);
        int addr = n * BS_STRIDE + kk;
        bs_hi[addr] = rh;
        bs_lo[addr] = rl;
    }
    __syncthreads();

    // ---- 4) GEMM: each wave computes 16(M) x 64(N) strips ------------------
    const int wave = tid >> 5;
    const int lane = tid & 31;
    const int half = lane >> 4;      // 0: K 0-7/16-23 ; 1: K 8-15/24-31
    const int mrow = lane & 15;      // A row within tile / B column

    for (int mt = wave; mt < 13; mt += 8) {
        int r  = mt * 16 + mrow;
        int rr = (r < 196) ? r : 0;              // clamp padding rows
        int l  = rr / 14, n = rr - l * 14;

        v8f acc[4];
        acc[0] = (v8f)0.f; acc[1] = (v8f)0.f; acc[2] = (v8f)0.f; acc[3] = (v8f)0.f;

        for (int ch = 0; ch < 12; ++ch) {        // K chunks of 32
            int tap = ch >> 2;
            int j0  = (ch & 3) * 32;
            int ab  = (l * 16 + n + tap) * 136 + j0 + half * 8;
            v16bf a_hi = ld_frag(xs_hi + ab, xs_hi + ab + 16);
            v16bf a_lo = ld_frag(xs_lo + ab, xs_lo + ab + 16);
            int kk0 = ch * 32 + half * 8;
            #pragma unroll
            for (int t = 0; t < 4; ++t) {
                int bb = (t * 16 + mrow) * BS_STRIDE + kk0;
                v16bf b_hi = ld_frag(bs_hi + bb, bs_hi + bb + 16);
                v16bf b_lo = ld_frag(bs_lo + bb, bs_lo + bb + 16);
                acc[t] = __builtin_amdgcn_wmma_f32_16x16x32_bf16(
                            false, a_hi, false, b_hi, (short)0, acc[t], false, false);
                acc[t] = __builtin_amdgcn_wmma_f32_16x16x32_bf16(
                            false, a_lo, false, b_hi, (short)0, acc[t], false, false);
                acc[t] = __builtin_amdgcn_wmma_f32_16x16x32_bf16(
                            false, a_hi, false, b_lo, (short)0, acc[t], false, false);
            }
        }

        // spill t4 tile to LDS (C layout: elem v -> M = v + half*8, N = mrow)
        #pragma unroll
        for (int t = 0; t < 4; ++t) {
            #pragma unroll
            for (int v = 0; v < 8; ++v) {
                int row = mt * 16 + half * 8 + v;
                if (row < 196) ts[row * TS_STRIDE + t * 16 + mrow] = acc[t][v];
            }
        }
    }
    __syncthreads();

    // ---- 5) epilogue: rolled combination with w1, coalesced store ----------
    const float* w1p = w1 + c0 * 2;
    float* outm = out + (size_t)m * 50176;
    for (int i = tid; i < 64 * 196; i += 256) {
        int cl = i / 196; int ab = i - cl * 196;
        int a = ab / 14; int b = ab - a * 14;
        int l1 = (a >= 1) ? (a - 1) : (a + 13);   // (a-1) mod 14
        int l2 = (a >= 2) ? (a - 2) : (a + 12);   // (a-2) mod 14
        float t1 = ts[(l1 * 14 + b) * TS_STRIDE + cl];
        float t2 = ts[(l2 * 14 + b) * TS_STRIDE + cl];
        outm[(c0 + cl) * 196 + ab] = t1 * w1p[cl * 2 + 0] + t2 * w1p[cl * 2 + 1];
    }
}

extern "C" void kernel_launch(void* const* d_in, const int* in_sizes, int n_in,
                              void* d_out, int out_size, void* d_ws, size_t ws_size,
                              hipStream_t stream) {
    (void)in_sizes; (void)n_in; (void)out_size; (void)d_ws; (void)ws_size;
    const float* x  = (const float*)d_in[0];   // (1024, 1792, 14)
    const float* w1 = (const float*)d_in[1];   // (256, 2)
    const float* w2 = (const float*)d_in[2];   // (128, 3, 256)
    float* out = (float*)d_out;                // (1024, 256, 14, 14)

    dim3 grid(4, 1024);   // (c-chunk of 64, m)
    dim3 block(256);      // 8 waves (wave32)
    fused_conv_roll_wmma<<<grid, block, SMEM_BYTES, stream>>>(x, w1, w2, out);
}
